// GCN_54039278519092
// MI455X (gfx1250) — matrix operand order repaired and checked by reading, not compiled, not run on hardware
//
#include <hip/hip_runtime.h>
#include <hip/hip_bf16.h>
#include <stdint.h>

// ---------------------------------------------------------------------------
// Types for CDNA5 WMMA (wave32): 16x16x32 bf16 -> f32
// ---------------------------------------------------------------------------
typedef __attribute__((ext_vector_type(16))) __bf16 v16bf;
typedef __attribute__((ext_vector_type(8)))  float  v8f;

// float -> bf16 with round-to-nearest-even (bit math; avoids relying on
// __bf16 fp-conversion lowering quirks)
static __device__ __forceinline__ __bf16 f2bf(float f) {
    union { float f; uint32_t u; } v; v.f = f;
    uint32_t r = v.u + 0x7FFFu + ((v.u >> 16) & 1u);
    uint16_t h = (uint16_t)(r >> 16);
    return __builtin_bit_cast(__bf16, h);
}

// ---------------------------------------------------------------------------
// Degree / normalization
// ---------------------------------------------------------------------------
__global__ void gcn_deg_fill(uint32_t* __restrict__ deg, int N) {
    int i = blockIdx.x * blockDim.x + threadIdx.x;
    if (i < N) deg[i] = 1u;            // self loop
}

__global__ void gcn_deg_count(const long long* __restrict__ dst,
                              uint32_t* __restrict__ deg, int E) {
    int e = blockIdx.x * blockDim.x + threadIdx.x;
    if (e < E) atomicAdd(&deg[(int)dst[e]], 1u);
}

__global__ void gcn_dinv(const uint32_t* __restrict__ deg,
                         float* __restrict__ dinv, int N) {
    int i = blockIdx.x * blockDim.x + threadIdx.x;
    if (i < N) dinv[i] = rsqrtf((float)deg[i]);
}

// ---------------------------------------------------------------------------
// Pack W (fp32 row-major [K][Nout]) into bf16 WMMA B-fragments:
//   Wp[((kt*NT + nt)*32 + lane)*16 + e] = W[kt*32 + (e<8 ? 8*(lane>>4)+e : 16+8*(lane>>4)+(e-8))][nt*16 + (lane&15)]
// (16-bit B 32x16 layout: lane n=lane&15, half=lane>>4, element e<8 <-> K=8*half+e, e>=8 <-> K=16+8*half+(e-8))
// ---------------------------------------------------------------------------
__global__ void gcn_pack_w(const float* __restrict__ W, __bf16* __restrict__ Wp,
                           int K, int Nout) {
    int NT = Nout >> 4;
    int idx = blockIdx.x * blockDim.x + threadIdx.x;
    int total = K * Nout;
    if (idx >= total) return;
    int e    = idx & 15;
    int lane = (idx >> 4) & 31;
    int frag = idx >> 9;
    int nt = frag % NT;
    int kt = frag / NT;
    int half = lane >> 4;
    int n    = lane & 15;
    int k    = kt * 32 + (e < 8 ? 8 * half + e : 16 + 8 * half + (e - 8));
    int col  = nt * 16 + n;
    Wp[idx] = f2bf(W[(size_t)k * Nout + col]);
}

// ---------------------------------------------------------------------------
// GEMM: H[N x (NT*16)] = act(X[N x K]) @ W[K x (NT*16)]  via v_wmma_f32_16x16x32_bf16
// One wave computes a 16-row strip across all NT column tiles (A reuse = NT).
// ---------------------------------------------------------------------------
template <int K, int NT, bool RELU>
__global__ void __launch_bounds__(256)
gcn_gemm_wmma(const float* __restrict__ X, const __bf16* __restrict__ Wp,
              float* __restrict__ H, int nrows) {
    const int lane = threadIdx.x & 31;
    const int wave = blockIdx.x * (blockDim.x >> 5) + (threadIdx.x >> 5);
    const int row0 = wave << 4;
    if (row0 >= nrows) return;          // wave-uniform: EXEC stays all-ones

    const int m    = lane & 15;
    const int half = lane >> 4;
    const float* xrow = X + (size_t)(row0 + m) * K;

    v8f acc[NT] = {};

#pragma unroll
    for (int kt = 0; kt < K / 32; ++kt) {
        const int k0 = kt * 32;
        // 16-bit A 16x32 layout: e<8 -> K = k0 + 8*half + e ; e>=8 -> K = k0+16+8*half+(e-8)
        const float4* p0 = (const float4*)(xrow + k0 + 8 * half);        // 32B aligned
        const float4* p1 = (const float4*)(xrow + k0 + 16 + 8 * half);   // 32B aligned
        float4 q0 = p0[0], q1 = p0[1], q2 = p1[0], q3 = p1[1];
        if (RELU) {
            q0.x = fmaxf(q0.x, 0.f); q0.y = fmaxf(q0.y, 0.f);
            q0.z = fmaxf(q0.z, 0.f); q0.w = fmaxf(q0.w, 0.f);
            q1.x = fmaxf(q1.x, 0.f); q1.y = fmaxf(q1.y, 0.f);
            q1.z = fmaxf(q1.z, 0.f); q1.w = fmaxf(q1.w, 0.f);
            q2.x = fmaxf(q2.x, 0.f); q2.y = fmaxf(q2.y, 0.f);
            q2.z = fmaxf(q2.z, 0.f); q2.w = fmaxf(q2.w, 0.f);
            q3.x = fmaxf(q3.x, 0.f); q3.y = fmaxf(q3.y, 0.f);
            q3.z = fmaxf(q3.z, 0.f); q3.w = fmaxf(q3.w, 0.f);
        }
        v16bf a;
        a[0]  = f2bf(q0.x); a[1]  = f2bf(q0.y); a[2]  = f2bf(q0.z); a[3]  = f2bf(q0.w);
        a[4]  = f2bf(q1.x); a[5]  = f2bf(q1.y); a[6]  = f2bf(q1.z); a[7]  = f2bf(q1.w);
        a[8]  = f2bf(q2.x); a[9]  = f2bf(q2.y); a[10] = f2bf(q2.z); a[11] = f2bf(q2.w);
        a[12] = f2bf(q3.x); a[13] = f2bf(q3.y); a[14] = f2bf(q3.z); a[15] = f2bf(q3.w);

#pragma unroll
        for (int nt = 0; nt < NT; ++nt) {
            v16bf b = *(const v16bf*)(Wp + (((size_t)kt * NT + nt) * 32 + lane) * 16);
            acc[nt] = __builtin_amdgcn_wmma_f32_16x16x32_bf16(
                false, a, false, b, (short)0, acc[nt], false, false);
        }
    }

    // C/D layout: element r <-> M = 8*half + r, N = lane&15
    const int Nout = NT * 16;
    float* hbase = H + (size_t)(row0 + 8 * half) * Nout + m;
#pragma unroll
    for (int nt = 0; nt < NT; ++nt) {
#pragma unroll
        for (int r = 0; r < 8; ++r) {
            hbase[(size_t)r * Nout + nt * 16] = acc[nt][r];
        }
    }
}

// ---------------------------------------------------------------------------
// Aggregation: Out = dinv^2 * H + bias   (self loop + bias), then edge scatter
// ---------------------------------------------------------------------------
__global__ void gcn_agg_init(const float* __restrict__ H,
                             const float* __restrict__ dinv,
                             const float* __restrict__ bias,
                             float* __restrict__ Out, int N, int F) {
    size_t idx = (size_t)blockIdx.x * blockDim.x + threadIdx.x;
    size_t total = (size_t)N * F;
    if (idx >= total) return;
    int i = (int)(idx / F);
    int f = (int)(idx % F);
    float d = dinv[i];
    Out[idx] = d * d * H[idx] + bias[f];
}

template <int F>
__global__ void gcn_agg_edges(const long long* __restrict__ src,
                              const long long* __restrict__ dst,
                              const float* __restrict__ H,
                              const float* __restrict__ dinv,
                              float* __restrict__ Out, int E) {
    constexpr int CH = F / 4;                 // float4 chunks per row
    size_t idx = (size_t)blockIdx.x * blockDim.x + threadIdx.x;
    size_t total = (size_t)E * CH;
    if (idx >= total) return;
    int e = (int)(idx / CH);
    int c = (int)(idx % CH);
    int s = (int)src[e];
    int d = (int)dst[e];
    float nrm = dinv[s] * dinv[d];
    float4 hv = *(const float4*)(H + (size_t)s * F + 4 * c);
    float* op = Out + (size_t)d * F + 4 * c;
    atomicAdd(op + 0, nrm * hv.x);
    atomicAdd(op + 1, nrm * hv.y);
    atomicAdd(op + 2, nrm * hv.z);
    atomicAdd(op + 3, nrm * hv.w);
}

// ---------------------------------------------------------------------------
// Host-side launcher
// ---------------------------------------------------------------------------
static inline size_t align256(size_t x) { return (x + 255) & ~(size_t)255; }
static inline unsigned cdiv(long long a, long long b) {
    return (unsigned)((a + b - 1) / b);
}

extern "C" void kernel_launch(void* const* d_in, const int* in_sizes, int n_in,
                              void* d_out, int out_size, void* d_ws, size_t ws_size,
                              hipStream_t stream) {
    const int Cin = 128, Ch = 128, Cout = 64;
    const int N = in_sizes[0] / Cin;
    const int E = in_sizes[1] / 2;

    const float*     x  = (const float*)d_in[0];
    const long long* ei = (const long long*)d_in[1];   // int64 edge_index [2, E]
    const float*     W1 = (const float*)d_in[2];
    const float*     b1 = (const float*)d_in[3];
    const float*     W2 = (const float*)d_in[4];
    const float*     b2 = (const float*)d_in[5];
    float*           out = (float*)d_out;

    const long long* src = ei;
    const long long* dst = ei + E;

    // workspace partitioning
    char* ws = (char*)d_ws;
    size_t off = 0;
    uint32_t* deg  = (uint32_t*)(ws + off); off = align256(off + (size_t)N * 4);
    float*    dinv = (float*)(ws + off);    off = align256(off + (size_t)N * 4);
    __bf16*   W1p  = (__bf16*)(ws + off);   off = align256(off + (size_t)Cin * Ch * 2);
    __bf16*   W2p  = (__bf16*)(ws + off);   off = align256(off + (size_t)Ch * Cout * 2);
    float*    h1   = (float*)(ws + off);    off = align256(off + (size_t)N * Ch * 4);
    float*    out1 = (float*)(ws + off);    off = align256(off + (size_t)N * Ch * 4);
    float*    h2   = (float*)(ws + off);    off = align256(off + (size_t)N * Cout * 4);
    (void)ws_size; (void)n_in; (void)out_size;

    const int B = 256;

    // 1) symmetric normalization coefficients
    gcn_deg_fill<<<cdiv(N, B), B, 0, stream>>>(deg, N);
    gcn_deg_count<<<cdiv(E, B), B, 0, stream>>>(dst, deg, E);
    gcn_dinv<<<cdiv(N, B), B, 0, stream>>>(deg, dinv, N);

    // 2) pack weights into WMMA bf16 B-fragment layout
    gcn_pack_w<<<cdiv((long long)Cin * Ch, B), B, 0, stream>>>(W1, W1p, Cin, Ch);
    gcn_pack_w<<<cdiv((long long)Ch * Cout, B), B, 0, stream>>>(W2, W2p, Ch, Cout);

    const int wavesPerBlock = B / 32;
    const unsigned gemmBlocks = cdiv((long long)(N / 16), wavesPerBlock);

    // 3) layer 1: h1 = x @ W1 (WMMA), out1 = norm-agg(h1) + b1
    gcn_gemm_wmma<128, 8, false><<<gemmBlocks, B, 0, stream>>>(x, W1p, h1, N);
    gcn_agg_init<<<cdiv((long long)N * Ch, B), B, 0, stream>>>(h1, dinv, b1, out1, N, Ch);
    gcn_agg_edges<128><<<cdiv((long long)E * (Ch / 4), B), B, 0, stream>>>(
        src, dst, h1, dinv, out1, E);

    // 4) layer 2: h2 = relu(out1) @ W2 (relu fused into A-load), out = norm-agg(h2) + b2
    gcn_gemm_wmma<128, 4, true><<<gemmBlocks, B, 0, stream>>>(out1, W2p, h2, N);
    gcn_agg_init<<<cdiv((long long)N * Cout, B), B, 0, stream>>>(h2, dinv, b2, out, N, Cout);
    gcn_agg_edges<64><<<cdiv((long long)E * (Cout / 4), B), B, 0, stream>>>(
        src, dst, h2, dinv, out, E);
}
